// Cdist_34952443855259
// MI455X (gfx1250) — matrix operand order repaired and checked by reading, not compiled, not run on hardware
//
#include <hip/hip_runtime.h>
#include <hip/hip_bf16.h>

// ---------------------------------------------------------------------------
// cdist mean, fused:  mean( sqrt(max(x2_i + y2_j - 2*x_i.y_j, 0)) )
// T = 8192, K = 512.  Pipeline: fp32->bf16 pre-convert, async global->LDS
// tile staging (double buffered, ASYNCcnt), bf16 WMMA 16x16x32 f32-accum,
// fused sqrt + deterministic reduction.
// ---------------------------------------------------------------------------

#define T_ROWS   8192
#define K_DIM    512
#define BM       128
#define BN       128
#define KC       32          // one WMMA K-step per chunk
#define LDA      40          // padded LDS row stride in bf16 elems (80B, conflict-free)
#define NKT      (K_DIM / KC)
#define NPART    (64 * 64)

typedef __attribute__((ext_vector_type(4)))  __bf16 bf16x4;
typedef __attribute__((ext_vector_type(8)))  __bf16 bf16x8;
typedef __attribute__((ext_vector_type(16))) __bf16 bf16x16;
typedef __attribute__((ext_vector_type(8)))  float  f32x8;

static __device__ __forceinline__ float wave_reduce_add(float v) {
    v += __shfl_xor(v, 16, 32);
    v += __shfl_xor(v, 8, 32);
    v += __shfl_xor(v, 4, 32);
    v += __shfl_xor(v, 2, 32);
    v += __shfl_xor(v, 1, 32);
    return v;
}

// low 32 bits of a generic pointer into LDS == byte offset within the
// workgroup's LDS allocation
static __device__ __forceinline__ unsigned int lds_off_u32(const void* p) {
    return (unsigned int)(unsigned long long)(uintptr_t)p;
}

// ASYNCcnt-tracked 16B global -> LDS copy (no VGPR data path)
static __device__ __forceinline__ void async_copy_b128(unsigned int lds_off,
                                                       const void* gptr) {
    asm volatile("global_load_async_to_lds_b128 %0, %1, off"
                 :: "v"(lds_off),
                    "v"((unsigned long long)(uintptr_t)gptr)
                 : "memory");
}

static __device__ __forceinline__ void wait_async0() {
#if __has_builtin(__builtin_amdgcn_s_wait_asynccnt)
    __builtin_amdgcn_s_wait_asynccnt(0);
#else
    asm volatile("s_wait_asynccnt 0x0" ::: "memory");
#endif
}

// -------------------- fp32 -> bf16 pre-convert -----------------------------
__global__ void cdist_cvt_kernel(const float* __restrict__ src,
                                 __bf16* __restrict__ dst) {
    const size_t i = (size_t)blockIdx.x * 256 + threadIdx.x;   // float4 index
    const float4 v = ((const float4*)src)[i];
    bf16x4 b;
    b[0] = (__bf16)v.x; b[1] = (__bf16)v.y;
    b[2] = (__bf16)v.z; b[3] = (__bf16)v.w;
    ((bf16x4*)dst)[i] = b;
}

// -------------------- row norms (fp32, exact) ------------------------------
__global__ void cdist_norms_kernel(const float* __restrict__ x,
                                   const float* __restrict__ y,
                                   float* __restrict__ norms) {
    const int row  = blockIdx.x * 8 + (threadIdx.x >> 5);
    const int lane = threadIdx.x & 31;
    const float* src = (row < T_ROWS) ? (x + (size_t)row * K_DIM)
                                      : (y + (size_t)(row - T_ROWS) * K_DIM);
    float s = 0.0f;
#pragma unroll
    for (int q = 0; q < 4; ++q) {
        const float4 v = *(const float4*)(src + q * 128 + lane * 4);
        s = fmaf(v.x, v.x, s);
        s = fmaf(v.y, v.y, s);
        s = fmaf(v.z, v.z, s);
        s = fmaf(v.w, v.w, s);
    }
    s = wave_reduce_add(s);
    if (lane == 0) norms[row] = s;
}

// -------------------- main fused tile kernel -------------------------------
__global__ void __launch_bounds__(256)
cdist_tile_kernel(const __bf16* __restrict__ xb,
                  const __bf16* __restrict__ yb,
                  const float* __restrict__ x2,
                  const float* __restrict__ y2,
                  float* __restrict__ partials) {
    __shared__ __bf16 ldsA[2][BM * LDA];
    __shared__ __bf16 ldsB[2][BN * LDA];
    __shared__ float  ldsX2[BM];
    __shared__ float  ldsY2[BN];
    __shared__ float  ldsPart[8];

    const int t      = threadIdx.x;
    const int wave   = t >> 5;
    const int lane   = t & 31;
    const int wave_m = wave >> 1;        // 0..3 -> 32 rows each
    const int wave_n = wave & 1;         // 0..1 -> 64 cols each
    const int m16    = lane & 15;
    const int half   = lane >> 4;

    const int i0 = blockIdx.y * BM;      // x-row base (M)
    const int j0 = blockIdx.x * BN;      // y-row base (N)

    if (t < 128) ldsX2[t] = x2[i0 + t];
    else         ldsY2[t - 128] = y2[j0 + (t - 128)];

    // --- per-thread async staging addresses: lane copies 16B (8 bf16) ------
    const int srow = t >> 2;             // 0..63 (row within 64-row pass)
    const int sseg = (t & 3) * 16;       // 16B segment within a 64B K-chunk row
    const char* gA0 = (const char*)(xb + (size_t)(i0 + srow)      * K_DIM) + sseg;
    const char* gA1 = (const char*)(xb + (size_t)(i0 + 64 + srow) * K_DIM) + sseg;
    const char* gB0 = (const char*)(yb + (size_t)(j0 + srow)      * K_DIM) + sseg;
    const char* gB1 = (const char*)(yb + (size_t)(j0 + 64 + srow) * K_DIM) + sseg;
    unsigned int lA0[2], lA1[2], lB0[2], lB1[2];
#pragma unroll
    for (int p = 0; p < 2; ++p) {
        lA0[p] = lds_off_u32(&ldsA[p][0]) + srow * (LDA * 2) + sseg;
        lA1[p] = lds_off_u32(&ldsA[p][0]) + (64 + srow) * (LDA * 2) + sseg;
        lB0[p] = lds_off_u32(&ldsB[p][0]) + srow * (LDA * 2) + sseg;
        lB1[p] = lds_off_u32(&ldsB[p][0]) + (64 + srow) * (LDA * 2) + sseg;
    }

    f32x8 acc[2][4];
#pragma unroll
    for (int sm = 0; sm < 2; ++sm)
#pragma unroll
        for (int sn = 0; sn < 4; ++sn)
#pragma unroll
            for (int i = 0; i < 8; ++i) acc[sm][sn][i] = 0.0f;

    // prologue: stage chunk 0 into buffer 0
    {
        async_copy_b128(lA0[0], gA0);
        async_copy_b128(lA1[0], gA1);
        async_copy_b128(lB0[0], gB0);
        async_copy_b128(lB1[0], gB1);
    }

    for (int kt = 0; kt < NKT; ++kt) {
        const int p = kt & 1;
        wait_async0();                   // own async copies for buf p done
        __syncthreads();                 // everyone's copies done, prev reads done

        if (kt + 1 < NKT) {              // overlap next chunk copy with WMMAs
            const size_t kB = (size_t)(kt + 1) * (KC * 2);   // byte advance
            async_copy_b128(lA0[p ^ 1], gA0 + kB);
            async_copy_b128(lA1[p ^ 1], gA1 + kB);
            async_copy_b128(lB0[p ^ 1], gB0 + kB);
            async_copy_b128(lB1[p ^ 1], gB1 + kB);
        }

        const __bf16* Ab = ldsA[p];
        const __bf16* Bb = ldsB[p];

        bf16x16 afrag[2];
#pragma unroll
        for (int sm = 0; sm < 2; ++sm) {
            const int base = (wave_m * 32 + sm * 16 + m16) * LDA;
            const bf16x8 lo = *(const bf16x8*)(Ab + base + half * 8);       // K=h*8+0..7
            const bf16x8 hi = *(const bf16x8*)(Ab + base + 16 + half * 8);  // K=16+h*8+0..7
            afrag[sm] = __builtin_shufflevector(lo, hi,
                0,1,2,3,4,5,6,7,8,9,10,11,12,13,14,15);
        }
        bf16x16 bfrag[4];
#pragma unroll
        for (int sn = 0; sn < 4; ++sn) {
            const int base = (wave_n * 64 + sn * 16 + m16) * LDA;
            const bf16x8 lo = *(const bf16x8*)(Bb + base + half * 8);       // K=h*8+0..7
            const bf16x8 hi = *(const bf16x8*)(Bb + base + 16 + half * 8);  // K=16+h*8+0..7
            bfrag[sn] = __builtin_shufflevector(lo, hi,
                0,1,2,3,4,5,6,7,8,9,10,11,12,13,14,15);
        }

#pragma unroll
        for (int sm = 0; sm < 2; ++sm)
#pragma unroll
            for (int sn = 0; sn < 4; ++sn)
                acc[sm][sn] = __builtin_amdgcn_wmma_f32_16x16x32_bf16(
                    false, afrag[sm], false, bfrag[sn],
                    (short)0, acc[sm][sn], false, false);
    }

    // ---- fused epilogue: sq = x2 + y2 - 2*dot ; sum(sqrt(max(sq,0))) ------
    float lsum = 0.0f;
#pragma unroll
    for (int sm = 0; sm < 2; ++sm) {
#pragma unroll
        for (int sn = 0; sn < 4; ++sn) {
            const float y2v = ldsY2[wave_n * 64 + sn * 16 + m16];
            const f32x8 a = acc[sm][sn];
#pragma unroll
            for (int v = 0; v < 8; ++v) {
                const float x2v = ldsX2[wave_m * 32 + sm * 16 + v + 8 * half];
                float s = fmaf(-2.0f, a[v], x2v + y2v);
                s = fmaxf(s, 0.0f);
                lsum += sqrtf(s);
            }
        }
    }

    lsum = wave_reduce_add(lsum);
    if (lane == 0) ldsPart[wave] = lsum;
    __syncthreads();
    if (t == 0) {
        float bs = 0.0f;
#pragma unroll
        for (int i = 0; i < 8; ++i) bs += ldsPart[i];
        partials[blockIdx.y * gridDim.x + blockIdx.x] = bs;
    }
}

// -------------------- final deterministic reduction ------------------------
__global__ void cdist_finalize_kernel(const float* __restrict__ partials,
                                      float* __restrict__ out) {
    __shared__ float ldsPart[8];
    const int t = threadIdx.x;
    float s = 0.0f;
    for (int i = t; i < NPART; i += 256) s += partials[i];   // fixed order
    s = wave_reduce_add(s);
    if ((t & 31) == 0) ldsPart[t >> 5] = s;
    __syncthreads();
    if (t == 0) {
        float total = 0.0f;
#pragma unroll
        for (int i = 0; i < 8; ++i) total += ldsPart[i];
        out[0] = total * (1.0f / (float)((double)T_ROWS * (double)T_ROWS));
    }
}

// ---------------------------------------------------------------------------
extern "C" void kernel_launch(void* const* d_in, const int* in_sizes, int n_in,
                              void* d_out, int out_size, void* d_ws, size_t ws_size,
                              hipStream_t stream) {
    (void)in_sizes; (void)n_in; (void)out_size; (void)ws_size;
    const float* x = (const float*)d_in[0];
    const float* y = (const float*)d_in[1];
    float* out = (float*)d_out;

    char* ws = (char*)d_ws;
    __bf16* xb      = (__bf16*)ws;                               // 8 MB
    __bf16* yb      = (__bf16*)(ws + (size_t)8 * 1024 * 1024);   // 8 MB
    float*  norms   = (float*)(ws + (size_t)16 * 1024 * 1024);   // 16384 floats
    float*  x2      = norms;
    float*  y2      = norms + T_ROWS;
    float*  partials = norms + 2 * T_ROWS;                       // 4096 floats

    const int n4 = (T_ROWS * K_DIM) / 4;                         // float4 count
    cdist_cvt_kernel<<<dim3(n4 / 256), dim3(256), 0, stream>>>(x, xb);
    cdist_cvt_kernel<<<dim3(n4 / 256), dim3(256), 0, stream>>>(y, yb);
    cdist_norms_kernel<<<dim3((2 * T_ROWS) / 8), dim3(256), 0, stream>>>(x, y, norms);
    cdist_tile_kernel<<<dim3(T_ROWS / BN, T_ROWS / BM), dim3(256), 0, stream>>>(
        xb, yb, x2, y2, partials);
    cdist_finalize_kernel<<<dim3(1), dim3(256), 0, stream>>>(partials, out);
}
